// MambaDecoderHead_61718680044225
// MI455X (gfx1250) — matrix operand order repaired and checked by reading, not compiled, not run on hardware
//
#include <hip/hip_runtime.h>

// ---------------------------------------------------------------------------
// MI455X (gfx1250) implementation: conv/projection GEMMs via bf16 WMMA
// (v_wmma_f32_16x16x32_bf16, f32 accum) with double-buffered async
// global->LDS tile loads; selective scan via chunked two-pass parallel
// linear recurrence (one state per lane, 16-lane groups).
// ---------------------------------------------------------------------------

typedef __attribute__((ext_vector_type(16))) __bf16 v16bf;
typedef __attribute__((ext_vector_type(8)))  float  v8f;

#define LDST  40      // LDS row stride in bf16 elems (80B: 16B-aligned, padded)
#define CHUNK 256     // scan chunk length

#if defined(__gfx1250__) && __has_builtin(__builtin_amdgcn_global_load_async_to_lds_b128)
#define USE_ASYNC_LDS 1
#else
#define USE_ASYNC_LDS 0
#endif

#if USE_ASYNC_LDS
// Builtin expects: (int4 addrspace(1)*, int4 addrspace(3)*, imm offset, imm cpol)
typedef int v4i_ __attribute__((__vector_size__(16)));
#define GAS1 __attribute__((address_space(1)))
#define LAS3 __attribute__((address_space(3)))
static __device__ __forceinline__ void async_copy_b128(const void* g, void* l) {
  __builtin_amdgcn_global_load_async_to_lds_b128(
      (GAS1 v4i_*)g, (LAS3 v4i_*)l, 0, 0);
}
static __device__ __forceinline__ void wait_async0() {
#if __has_builtin(__builtin_amdgcn_s_wait_asynccnt)
  __builtin_amdgcn_s_wait_asynccnt(0);
#else
  asm volatile("s_wait_asynccnt 0x0" ::: "memory");
#endif
}
#endif

// ------------------------------ workspace map ------------------------------
static constexpr size_t MBYTE     = 1ull << 20;
static constexpr size_t OFF_A     = 0;          // 64 MB  A / im2col bf16
static constexpr size_t OFF_B     = 64 * MBYTE; //  8 MB  B^T weights bf16
static constexpr size_t OFF_SEQI  = 72 * MBYTE; // 32 MB  h1/h2/h3 seq f32
static constexpr size_t OFF_SEQO  = 104 * MBYTE;// 32 MB  out1/out2 seq f32
static constexpr size_t OFF_XZ    = 136 * MBYTE;//128 MB  xz f32
static constexpr size_t OFF_UF    = 264 * MBYTE;// 64 MB  u f32
static constexpr size_t OFF_DBL   = 328 * MBYTE;// 16 MB  dt/B/C f32
static constexpr size_t OFF_DELTA = 344 * MBYTE;// 64 MB  delta f32
static constexpr size_t OFF_Y     = 408 * MBYTE;// 64 MB  scan y f32
static constexpr size_t OFF_SCAN  = 472 * MBYTE;// 16 MB  chunk states + Aneg

// ------------------------------ device helpers -----------------------------
static __device__ __forceinline__ float siluf_(float x) {
  return x / (1.f + __expf(-x));
}

// ------------------------------ im2col (bf16) ------------------------------
__global__ void im2col_bf16(const float* __restrict__ in, __bf16* __restrict__ out,
                            int layoutHWC, int Bn, int C, int H, int W,
                            int Hout, int Wout, int KH, int KW,
                            int stride, int pad, int Kpad)
{
  long long t = (long long)blockIdx.x * blockDim.x + threadIdx.x;
  long long tot = (long long)Bn * Hout * Wout * Kpad;
  if (t >= tot) return;
  int kk = (int)(t % Kpad);
  long long m = t / Kpad;
  int hw = (int)(m % ((long long)Hout * Wout));
  int b  = (int)(m / ((long long)Hout * Wout));
  int oh = hw / Wout, ow = hw % Wout;
  float v = 0.f;
  int Kr = C * KH * KW;
  if (kk < Kr) {
    int c  = kk / (KH * KW);
    int rr = kk % (KH * KW);
    int kh = rr / KW, kw = rr % KW;
    int ih = oh * stride - pad + kh;
    int iw = ow * stride - pad + kw;
    if (ih >= 0 && ih < H && iw >= 0 && iw < W) {
      v = layoutHWC
        ? in[((long long)b * H * W + (long long)ih * W + iw) * C + c]
        : in[(((long long)b * C + c) * H + ih) * (long long)W + iw];
    }
  }
  out[t] = (__bf16)v;
}

// ---------------- weight prep: (K,N) or (N,K) f32 -> (Npad,Kpad) bf16 ------
__global__ void prep_bt(const float* __restrict__ Wsrc, __bf16* __restrict__ Bt,
                        int Nreal, int Kreal, int Npad, int Kpad, int srcIsKN)
{
  long long t = (long long)blockIdx.x * blockDim.x + threadIdx.x;
  long long tot = (long long)Npad * Kpad;
  if (t >= tot) return;
  int k = (int)(t % Kpad);
  int n = (int)(t / Kpad);
  float v = 0.f;
  if (n < Nreal && k < Kreal)
    v = srcIsKN ? Wsrc[(long long)k * Nreal + n] : Wsrc[(long long)n * Kreal + k];
  Bt[t] = (__bf16)v;
}

// ------------------------------ WMMA GEMM ----------------------------------
// C[M x Nreal] = A[M x Kpad] * Bt[Npad x Kpad]^T   (bf16 in, f32 accum)
// modes: 0 = bias, 1 = bias+ReLU, 2 = bias+residual, 3 = bias + NCHW scatter
static __device__ __forceinline__ v16bf frag_a(const __bf16* As, int m0, int lane)
{
  int row = m0 + (lane & 15);
  int half = lane >> 4;
  const __bf16* p = As + row * LDST + half * 8;
  union { v16bf v; float4 q[2]; } u;
  u.q[0] = *reinterpret_cast<const float4*>(p);
  u.q[1] = *reinterpret_cast<const float4*>(p + 16);
  return u.v;
}
static __device__ __forceinline__ v16bf frag_b(const __bf16* Bs, int n0, int lane)
{
  int col = n0 + (lane & 15);
  int half = lane >> 4;
  const __bf16* p = Bs + col * LDST + half * 8;
  union { v16bf v; float4 q[2]; } u;
  u.q[0] = *reinterpret_cast<const float4*>(p);
  u.q[1] = *reinterpret_cast<const float4*>(p + 16);
  return u.v;
}

__global__ __launch_bounds__(256)
void gemm_bf16_wmma(const __bf16* __restrict__ A, const __bf16* __restrict__ Bt,
                    float* __restrict__ C, const float* __restrict__ bias,
                    const float* __restrict__ residual,
                    int Nreal, int Kpad, int mode, int HW)
{
  __shared__ __align__(16) __bf16 As[2][128 * LDST];
  __shared__ __align__(16) __bf16 Bs[2][128 * LDST];
  int tid = threadIdx.x;
  int lane = tid & 31, wid = tid >> 5;
  int waveM = wid & 3;        // 4 waves along M (32 rows each)
  int waveN = wid >> 2;       // 2 waves along N (64 cols each)
  long long rowA0 = (long long)blockIdx.x * 128;
  long long rowB0 = (long long)blockIdx.y * 128;

  v8f acc[2][4];
#pragma unroll
  for (int i = 0; i < 2; ++i)
#pragma unroll
    for (int j = 0; j < 4; ++j)
#pragma unroll
      for (int r = 0; r < 8; ++r) acc[i][j][r] = 0.f;

  int ldr = tid >> 2;         // 0..63
  int ldc = (tid & 3) * 8;    // 0,8,16,24

  // tile loader: 256 threads move 128x32 of A and of B^T (16B per thread-slot)
  auto issue_tile = [&](int p, int k0) {
#pragma unroll
    for (int rr = 0; rr < 128; rr += 64) {
      int row = ldr + rr;
      const __bf16* ga = A  + (rowA0 + row) * (long long)Kpad + k0 + ldc;
      const __bf16* gb = Bt + (rowB0 + row) * (long long)Kpad + k0 + ldc;
#if USE_ASYNC_LDS
      async_copy_b128(ga, &As[p][row * LDST + ldc]);
      async_copy_b128(gb, &Bs[p][row * LDST + ldc]);
#else
      *reinterpret_cast<float4*>(&As[p][row * LDST + ldc]) =
          *reinterpret_cast<const float4*>(ga);
      *reinterpret_cast<float4*>(&Bs[p][row * LDST + ldc]) =
          *reinterpret_cast<const float4*>(gb);
#endif
    }
  };

  int p = 0;
  issue_tile(0, 0);
#if USE_ASYNC_LDS
  wait_async0();
#endif
  __syncthreads();

  for (int k0 = 0; k0 < Kpad; k0 += 32) {
    if (k0 + 32 < Kpad) issue_tile(p ^ 1, k0 + 32);   // prefetch next K-tile

    v16bf bfr[4];
#pragma unroll
    for (int j = 0; j < 4; ++j) bfr[j] = frag_b(&Bs[p][0], waveN * 64 + j * 16, lane);
#pragma unroll
    for (int i = 0; i < 2; ++i) {
      v16bf afr = frag_a(&As[p][0], waveM * 32 + i * 16, lane);
#pragma unroll
      for (int j = 0; j < 4; ++j)
        acc[i][j] = __builtin_amdgcn_wmma_f32_16x16x32_bf16(
            false, afr, false, bfr[j], (short)0, acc[i][j], false, false);
    }

#if USE_ASYNC_LDS
    wait_async0();
#endif
    __syncthreads();
    p ^= 1;
  }

  // epilogue (C f32 layout per ISA 7.12.2: VGPR r -> M = base+r (+8 for hi half))
  int halfSel = (lane >> 4) * 8;
  int nl = lane & 15;
#pragma unroll
  for (int i = 0; i < 2; ++i)
#pragma unroll
    for (int j = 0; j < 4; ++j) {
      int nBase = (int)(blockIdx.y * 128) + waveN * 64 + j * 16 + nl;
      if (nBase >= Nreal) continue;
      long long mBase = rowA0 + waveM * 32 + i * 16 + halfSel;
      float bv = bias ? bias[nBase] : 0.f;
#pragma unroll
      for (int r = 0; r < 8; ++r) {
        long long m = mBase + r;
        float v = acc[i][j][r] + bv;
        if (mode == 1) v = v > 0.f ? v : 0.f;
        long long oidx;
        if (mode == 3) {
          long long b  = m / HW;
          long long hw = m - b * HW;
          oidx = (b * Nreal + nBase) * (long long)HW + hw;
        } else {
          oidx = m * (long long)Nreal + nBase;
          if (mode == 2) v += residual[oidx];
        }
        C[oidx] = v;
      }
    }
}

// ------------------------------ LayerNorm -> bf16 --------------------------
__global__ __launch_bounds__(256)
void ln_bf16(const float* __restrict__ x, const float* __restrict__ g,
             const float* __restrict__ b, __bf16* __restrict__ out,
             long long Mrows, int C)
{
  int lane = threadIdx.x & 31;
  long long row = (long long)blockIdx.x * (blockDim.x / 32) + (threadIdx.x >> 5);
  if (row >= Mrows) return;
  const float* xr = x + row * C;
  float s = 0.f, s2 = 0.f;
  for (int c = lane; c < C; c += 32) { float v = xr[c]; s += v; s2 += v * v; }
#pragma unroll
  for (int m = 16; m; m >>= 1) { s += __shfl_xor(s, m); s2 += __shfl_xor(s2, m); }
  float mean = s / C;
  float inv = rsqrtf(s2 / C - mean * mean + 1e-5f);
  for (int c = lane; c < C; c += 32)
    out[row * C + c] = (__bf16)((xr[c] - mean) * inv * g[c] + b[c]);
}

// --------------------- causal depthwise conv(K=3) + SiLU -------------------
__global__ void dwconv_silu(const float* __restrict__ xz, const float* __restrict__ w,
                            const float* __restrict__ bias, float* __restrict__ uf,
                            __bf16* __restrict__ ub, int Bn, int L, int di, int N2)
{
  long long t = (long long)blockIdx.x * blockDim.x + threadIdx.x;
  long long tot = (long long)Bn * L * di;
  if (t >= tot) return;
  int d = (int)(t % di);
  long long bl = t / di;
  int l = (int)(bl % L);
  long long b = bl / L;
  const float* base = xz + b * (long long)L * N2 + d;  // u = cols [0, di)
  float acc = bias[d];
#pragma unroll
  for (int k = 0; k < 3; ++k) {
    int ls = l - 2 + k;
    if (ls >= 0) acc += w[d * 3 + k] * base[(long long)ls * N2];
  }
  float v = siluf_(acc);
  uf[t] = v;
  ub[t] = (__bf16)v;
}

// ------------------- delta = softplus(dt @ W_dt + dt_b) --------------------
__global__ void delta_softplus(const float* __restrict__ dbl, const float* __restrict__ W_dt,
                               const float* __restrict__ dt_b, float* __restrict__ delta,
                               long long M, int di, int Ndbl, int dtRank)
{
  long long t = (long long)blockIdx.x * blockDim.x + threadIdx.x;
  if (t >= M * di) return;
  long long m = t / di;
  int d = (int)(t % di);
  float s = dt_b[d];
  for (int r = 0; r < dtRank; ++r) s += dbl[m * Ndbl + r] * W_dt[(long long)r * di + d];
  delta[t] = (s > 20.f) ? s : __logf(1.f + __expf(s));
}

__global__ void neg_exp(const float* __restrict__ a, float* __restrict__ o, int n)
{
  int i = blockIdx.x * blockDim.x + threadIdx.x;
  if (i < n) o[i] = -__expf(a[i]);
}

// ------------------------ chunked selective scan ---------------------------
// 16 lanes = one (b,d) channel, one state per lane.
// pass 0: zero-init chunk scan, record final state + prod(a) per chunk.
// pass 1: replay with true init (from combine), emit y via lane reduce.
__global__ __launch_bounds__(256)
void scan_pass(const float* __restrict__ delta, const float* __restrict__ u,
               const float* __restrict__ dbl, const float* __restrict__ Aneg,
               float* __restrict__ hlast, float* __restrict__ prods,
               const float* __restrict__ hinit, float* __restrict__ y,
               int Bn, int L, int di, int Ndbl, int dtRank, int pass)
{
  int n   = threadIdx.x & 15;
  int grp = threadIdx.x >> 4;
  int pair = blockIdx.y * 16 + grp;
  int b = pair / di, d = pair - b * di;
  int c = blockIdx.x;
  int l0 = c * CHUNK;
  int l1 = min(l0 + CHUNK, L);
  int nPairs = Bn * di;
  long long sidx = ((long long)c * nPairs + pair) * 16 + n;
  float a_coef = Aneg[d * 16 + n];
  float h = (pass == 1) ? hinit[sidx] : 0.f;
  float prod = 1.f;
  for (int l = l0; l < l1; ++l) {
    long long m = (long long)b * L + l;
    float dt = delta[m * di + d];
    float uu = u[m * di + d];
    float bv = dbl[m * Ndbl + dtRank + n];
    float a  = __expf(dt * a_coef);
    h = h * a + dt * uu * bv;
    if (pass == 0) {
      prod *= a;
    } else {
      float p = h * dbl[m * Ndbl + dtRank + 16 + n];
      p += __shfl_xor(p, 1);
      p += __shfl_xor(p, 2);
      p += __shfl_xor(p, 4);
      p += __shfl_xor(p, 8);
      if (n == 0) y[m * di + d] = p;
    }
  }
  if (pass == 0) { hlast[sidx] = h; prods[sidx] = prod; }
}

__global__ void scan_combine(const float* __restrict__ hlast, const float* __restrict__ prods,
                             float* __restrict__ hinit, long long nP16, int numChunks)
{
  long long i = (long long)blockIdx.x * blockDim.x + threadIdx.x;
  if (i >= nP16) return;
  float run = 0.f;
  for (int c = 0; c < numChunks; ++c) {
    long long idx = (long long)c * nP16 + i;
    hinit[idx] = run;
    run = run * prods[idx] + hlast[idx];
  }
}

// ----------------- gate: (y + u*D) * silu(z) -> bf16 A matrix --------------
__global__ void gate_mul(const float* __restrict__ y, const float* __restrict__ u,
                         const float* __restrict__ xz, const float* __restrict__ Dskip,
                         __bf16* __restrict__ out, long long M, int di, int N2)
{
  long long t = (long long)blockIdx.x * blockDim.x + threadIdx.x;
  if (t >= M * di) return;
  long long m = t / di;
  int d = (int)(t % di);
  float z = xz[m * N2 + di + d];
  out[t] = (__bf16)((y[t] + u[t] * Dskip[d]) * siluf_(z));
}

// ------------------------------ host helpers -------------------------------
static inline void launch_gemm(hipStream_t s, const __bf16* A, const __bf16* Bt, float* C,
                               const float* bias, const float* res,
                               int M, int Nreal, int Npad, int Kpad, int mode, int HW)
{
  dim3 g(M / 128, Npad / 128);
  gemm_bf16_wmma<<<g, 256, 0, s>>>(A, Bt, C, bias, res, Nreal, Kpad, mode, HW);
}

static void run_conv(hipStream_t s, const float* in, int layoutHWC,
                     int Bn, int C, int H, int W, int KH, int KW, int stride, int pad,
                     int Cout, const float* wgt, const float* bias,
                     float* out, int mode, char* ws)
{
  int Hout = (H + 2 * pad - KH) / stride + 1;
  int Wout = (W + 2 * pad - KW) / stride + 1;
  int Kr = C * KH * KW;
  int Kpad = ((Kr + 31) / 32) * 32;
  long long M = (long long)Bn * Hout * Wout;
  __bf16* Abuf = (__bf16*)(ws + OFF_A);
  __bf16* Bbuf = (__bf16*)(ws + OFF_B);
  long long tot = M * Kpad;
  im2col_bf16<<<(int)((tot + 255) / 256), 256, 0, s>>>(
      in, Abuf, layoutHWC, Bn, C, H, W, Hout, Wout, KH, KW, stride, pad, Kpad);
  int Npad = ((Cout + 127) / 128) * 128;
  long long totb = (long long)Npad * Kpad;
  prep_bt<<<(int)((totb + 255) / 256), 256, 0, s>>>(wgt, Bbuf, Cout, Kr, Npad, Kpad, 0);
  launch_gemm(s, Abuf, Bbuf, out, bias, nullptr, (int)M, Cout, Npad, Kpad, mode, Hout * Wout);
}

static void run_mamba(hipStream_t s, const float* seq, float* out_seq,
                      int Bn, int L, int dim, const float* const* mp, char* ws)
{
  int di = 2 * dim, dtRank = dim / 16, Ndbl = dtRank + 32, N2 = 2 * di;
  long long M = (long long)Bn * L;
  const float *ln_g = mp[0], *ln_b = mp[1], *W_in = mp[2], *b_in = mp[3],
              *conv_w = mp[4], *conv_b = mp[5], *W_x = mp[6], *W_dt = mp[7],
              *dt_b = mp[8], *A_log = mp[9], *D_skip = mp[10],
              *W_out = mp[11], *b_out = mp[12];
  __bf16* Abuf = (__bf16*)(ws + OFF_A);
  __bf16* Bbuf = (__bf16*)(ws + OFF_B);
  float* xz    = (float*)(ws + OFF_XZ);
  float* uf    = (float*)(ws + OFF_UF);
  float* dbl   = (float*)(ws + OFF_DBL);
  float* delta = (float*)(ws + OFF_DELTA);
  float* yv    = (float*)(ws + OFF_Y);
  float* hlast = (float*)(ws + OFF_SCAN);
  float* prods = (float*)(ws + OFF_SCAN + 5 * MBYTE);
  float* hinit = (float*)(ws + OFF_SCAN + 10 * MBYTE);
  float* Aneg  = (float*)(ws + OFF_SCAN + 15 * MBYTE);

  // LN(seq) -> bf16 A
  ln_bf16<<<(int)((M + 7) / 8), 256, 0, s>>>(seq, ln_g, ln_b, Abuf, M, dim);
  // xz = LN(seq) @ W_in + b_in
  int NpadIn = ((N2 + 127) / 128) * 128;
  prep_bt<<<(int)(((long long)NpadIn * dim + 255) / 256), 256, 0, s>>>(
      W_in, Bbuf, N2, dim, NpadIn, dim, 1);
  launch_gemm(s, Abuf, Bbuf, xz, b_in, nullptr, (int)M, N2, NpadIn, dim, 0, 0);
  // u = silu(dwconv(u)); also bf16 copy for next GEMM
  long long tMd = M * di;
  dwconv_silu<<<(int)((tMd + 255) / 256), 256, 0, s>>>(xz, conv_w, conv_b, uf, Abuf, Bn, L, di, N2);
  // dbl = u @ W_x
  int NpadX = ((Ndbl + 127) / 128) * 128;
  prep_bt<<<(int)(((long long)NpadX * di + 255) / 256), 256, 0, s>>>(
      W_x, Bbuf, Ndbl, di, NpadX, di, 1);
  launch_gemm(s, Abuf, Bbuf, dbl, nullptr, nullptr, (int)M, Ndbl, NpadX, di, 0, 0);
  // delta = softplus(dt @ W_dt + dt_b)
  delta_softplus<<<(int)((tMd + 255) / 256), 256, 0, s>>>(dbl, W_dt, dt_b, delta, M, di, Ndbl, dtRank);
  // A = -exp(A_log)
  neg_exp<<<(di * 16 + 255) / 256, 256, 0, s>>>(A_log, Aneg, di * 16);
  // chunked scan: pass0 -> combine -> pass1
  int numChunks = (L + CHUNK - 1) / CHUNK;
  int nPairs = Bn * di;
  dim3 sg(numChunks, nPairs / 16);
  scan_pass<<<sg, 256, 0, s>>>(delta, uf, dbl, Aneg, hlast, prods, hinit, yv,
                               Bn, L, di, Ndbl, dtRank, 0);
  long long nP16 = (long long)nPairs * 16;
  scan_combine<<<(int)((nP16 + 255) / 256), 256, 0, s>>>(hlast, prods, hinit, nP16, numChunks);
  scan_pass<<<sg, 256, 0, s>>>(delta, uf, dbl, Aneg, hlast, prods, hinit, yv,
                               Bn, L, di, Ndbl, dtRank, 1);
  // gated = (y + u*D) * silu(z) -> bf16 A
  gate_mul<<<(int)((tMd + 255) / 256), 256, 0, s>>>(yv, uf, xz, D_skip, Abuf, M, di, N2);
  // out = seq + gated @ W_out + b_out
  int NpadO = ((dim + 127) / 128) * 128;
  prep_bt<<<(int)(((long long)NpadO * di + 255) / 256), 256, 0, s>>>(
      W_out, Bbuf, dim, di, NpadO, di, 1);
  launch_gemm(s, Abuf, Bbuf, out_seq, b_out, seq, (int)M, dim, NpadO, di, 2, 0);
}

// ------------------------------- entry point -------------------------------
extern "C" void kernel_launch(void* const* d_in, const int* in_sizes, int n_in,
                              void* d_out, int out_size, void* d_ws, size_t ws_size,
                              hipStream_t stream)
{
  (void)in_sizes; (void)n_in; (void)out_size; (void)ws_size;
  const float* x  = (const float*)d_in[0];
  const float* w1 = (const float*)d_in[1]; const float* b1 = (const float*)d_in[2];
  const float* w2 = (const float*)d_in[3]; const float* b2 = (const float*)d_in[4];
  const float* w3 = (const float*)d_in[5]; const float* b3 = (const float*)d_in[6];
  const float* wh = (const float*)d_in[7]; const float* bh = (const float*)d_in[8];
  const float* m1[13]; for (int i = 0; i < 13; ++i) m1[i] = (const float*)d_in[9 + i];
  const float* m2[13]; for (int i = 0; i < 13; ++i) m2[i] = (const float*)d_in[22 + i];
  char* ws = (char*)d_ws;
  float* seqI = (float*)(ws + OFF_SEQI);
  float* seqO = (float*)(ws + OFF_SEQO);

  // conv1: NCHW (4,3,256,256) -> h1 seq (4, 16384, 128), ReLU
  run_conv(stream, x, 0, 4, 3, 256, 256, 4, 4, 2, 1, 128, w1, b1, seqI, 1, ws);
  run_mamba(stream, seqI, seqO, 4, 128 * 128, 128, m1, ws);
  // conv2 on HWC seq -> h2 seq (4, 4096, 256), ReLU
  run_conv(stream, seqO, 1, 4, 128, 128, 128, 4, 4, 2, 1, 256, w2, b2, seqI, 1, ws);
  run_mamba(stream, seqI, seqO, 4, 64 * 64, 256, m2, ws);
  // conv3 -> h3 seq (4, 1024, 512), ReLU
  run_conv(stream, seqO, 1, 4, 256, 64, 64, 4, 4, 2, 1, 512, w3, b3, seqI, 1, ws);
  // head 1x1 -> d_out as NCHW (4, 4096, 32, 32) == (4, 4, 1024, 1024)
  run_conv(stream, seqI, 1, 4, 512, 32, 32, 1, 1, 1, 0, 4096, wh, bh, (float*)d_out, 3, ws);
}
